// PiNet_75402445848831
// MI455X (gfx1250) — matrix operand order repaired and checked by reading, not compiled, not run on hardware
//
#include <hip/hip_runtime.h>

// ---------------------------------------------------------------------------
// PiNet degree-3 polynomial network for MI455X (gfx1250, wave32, WMMA).
// out = bias + x W1^T + (x(x)x) W2^T + (x(x)x(x)x) W3^T
// Roofline: W3 = 512 MB f32 read once = ~22 us @ 23.3 TB/s; bf16 WMMA
// 16x16x32 keeps compute under that floor.  Each wave owns 16 output cols x
// all 256 batch rows in registers (16 v8f accums), K split across grid.y so
// every W3 element is fetched exactly once.  Kronecker factor handled as
// diag(s) * (X Wij^T): raw bf16 x is the WMMA A operand (pure ds_load_b128),
// s scales the 16x16 result with 4 pk_fma per WMMA pair, software-pipelined
// one batch-tile behind the WMMAs to absorb the WMMA->VALU hazard.
// ---------------------------------------------------------------------------

typedef __bf16 bf16_t;
typedef __attribute__((ext_vector_type(16))) __bf16 v16bf;
typedef __attribute__((ext_vector_type(8)))  __bf16 v8bf;
typedef __attribute__((ext_vector_type(8)))  float  v8f;

#define BATCH   256
#define INDIM   64
#define OUTDIM  512
#define KSLICES 32          // grid.y : K split so W3 is read exactly once
#define WAVES   4
#define THREADS (WAVES * 32)
#define XSTRIDE 72          // padded bf16 row stride (36 dwords) -> no LDS bank conflicts
#define JC      16          // j-blocks whose scale vectors are staged per barrier

// f32x16 -> bf16x16 (compiler emits v_cvt_pk_bf16_f32)
__device__ __forceinline__ v16bf cvt_b(const float4* f) {
  v16bf b;
  b[0]=(bf16_t)f[0].x;  b[1]=(bf16_t)f[0].y;  b[2]=(bf16_t)f[0].z;  b[3]=(bf16_t)f[0].w;
  b[4]=(bf16_t)f[1].x;  b[5]=(bf16_t)f[1].y;  b[6]=(bf16_t)f[1].z;  b[7]=(bf16_t)f[1].w;
  b[8]=(bf16_t)f[2].x;  b[9]=(bf16_t)f[2].y;  b[10]=(bf16_t)f[2].z; b[11]=(bf16_t)f[2].w;
  b[12]=(bf16_t)f[3].x; b[13]=(bf16_t)f[3].y; b[14]=(bf16_t)f[3].z; b[15]=(bf16_t)f[3].w;
  return b;
}

// A operand (16-bit A 16x32 layout): two 16B ds_load_b128, zero VALU.
__device__ __forceinline__ v16bf load_a(const bf16_t* xrow, int h) {
  v8bf lo = *(const v8bf*)(xrow + 8 * h);
  v8bf hi = *(const v8bf*)(xrow + 16 + 8 * h);
  return __builtin_shufflevector(lo, hi, 0,1,2,3,4,5,6,7,8,9,10,11,12,13,14,15);
}

// One 64-wide K block: G = X_tile * Wblk^T (2 WMMA / batch tile), then
// acc += diag(s) * G, retired one batch-tile late to cover WMMA->VALU hazard.
__device__ __forceinline__ void do_block(const float* __restrict__ Wrow,
                                         const float* __restrict__ Wnext,
                                         const bf16_t* __restrict__ xb,
                                         const float* __restrict__ sb,
                                         int lane, v8f acc[16]) {
  const int h  = lane >> 4;
  const int ml = lane & 15;

  float4 w0[4], w1[4];
  {
    const float4* p0 = (const float4*)(Wrow + 8 * h);        // K 8h..8h+7 (+16) of 0..31
    const float4* p1 = (const float4*)(Wrow + 32 + 8 * h);   // K 8h..8h+7 (+16) of 32..63
    w0[0]=p0[0]; w0[1]=p0[1]; w0[2]=p0[4]; w0[3]=p0[5];
    w1[0]=p1[0]; w1[1]=p1[1]; w1[2]=p1[4]; w1[3]=p1[5];
  }
  if (Wnext) {  // pull both 128B lines of next block's per-lane region into L0
    __builtin_prefetch(Wnext + 16 * h, 0, 0);
    __builtin_prefetch(Wnext + 32 + 16 * h, 0, 0);
  }
  const v16bf b0 = cvt_b(w0);
  const v16bf b1 = cvt_b(w1);

  v8f   gp = {0,0,0,0,0,0,0,0};
  float4 sp0 = {0,0,0,0}, sp1 = {0,0,0,0};
#pragma unroll
  for (int bt = 0; bt < 16; ++bt) {
    const bf16_t* xr = xb + (bt * 16 + ml) * XSTRIDE;
    const v16bf a0 = load_a(xr, h);
    const v16bf a1 = load_a(xr + 32, h);
    v8f g = {0,0,0,0,0,0,0,0};
    g = __builtin_amdgcn_wmma_f32_16x16x32_bf16(false, a0, false, b0,
                                                (short)0, g, false, false);
    g = __builtin_amdgcn_wmma_f32_16x16x32_bf16(false, a1, false, b1,
                                                (short)0, g, false, false);
    const float4 s0 = *(const float4*)(sb + bt * 16 + 8 * h);
    const float4 s1 = *(const float4*)(sb + bt * 16 + 8 * h + 4);
    if (bt > 0) {  // retire previous tile: D layout row M = v + 8h
      acc[bt-1][0] += sp0.x * gp[0];
      acc[bt-1][1] += sp0.y * gp[1];
      acc[bt-1][2] += sp0.z * gp[2];
      acc[bt-1][3] += sp0.w * gp[3];
      acc[bt-1][4] += sp1.x * gp[4];
      acc[bt-1][5] += sp1.y * gp[5];
      acc[bt-1][6] += sp1.z * gp[6];
      acc[bt-1][7] += sp1.w * gp[7];
    }
    gp = g; sp0 = s0; sp1 = s1;
  }
  acc[15][0] += sp0.x * gp[0];
  acc[15][1] += sp0.y * gp[1];
  acc[15][2] += sp0.z * gp[2];
  acc[15][3] += sp0.w * gp[3];
  acc[15][4] += sp1.x * gp[4];
  acc[15][5] += sp1.y * gp[5];
  acc[15][6] += sp1.z * gp[6];
  acc[15][7] += sp1.w * gp[7];
}

__global__ void __launch_bounds__(THREADS)
pinet_main(const float* __restrict__ x, const float* __restrict__ W1,
           const float* __restrict__ W2, const float* __restrict__ W3,
           float* __restrict__ out) {
  __shared__ __align__(16) bf16_t xb[BATCH * XSTRIDE];  // 36 KB padded bf16 x
  __shared__ __align__(16) float  sall[JC * BATCH];     // 16 KB scale vectors

  const int tid  = threadIdx.x;
  const int lane = tid & 31;
  const int wave = tid >> 5;
  const int y    = blockIdx.y;                                 // K slice
  const int n    = blockIdx.x * 64 + wave * 16 + (lane & 15);  // output col

  for (int idx = tid; idx < BATCH * INDIM; idx += THREADS)
    xb[(idx >> 6) * XSTRIDE + (idx & 63)] = (bf16_t)x[idx];
  __syncthreads();

  v8f acc[16];
#pragma unroll
  for (int bt = 0; bt < 16; ++bt)
    acc[bt] = (v8f){0,0,0,0,0,0,0,0};

  // ---- degree 3: this slice owns i in {2y, 2y+1}, all j -------------------
  for (int ii = 0; ii < 2; ++ii) {
    const int i = 2 * y + ii;
    const float* Wib = W3 + (size_t)n * (INDIM * INDIM * INDIM)
                          + (size_t)i * (INDIM * INDIM);
    for (int jc = 0; jc < INDIM / JC; ++jc) {
      __syncthreads();
      for (int idx = tid; idx < JC * BATCH; idx += THREADS) {
        const int jj = idx >> 8, m = idx & 255;
        sall[idx] = x[m * INDIM + i] * x[m * INDIM + jc * JC + jj];
      }
      __syncthreads();
#pragma unroll 1
      for (int jj = 0; jj < JC; ++jj) {
        const float* cur = Wib + (jc * JC + jj) * INDIM;
        do_block(cur, cur + INDIM, xb, sall + jj * BATCH, lane, acc);
      }
    }
  }

  // ---- degree 2: this slice owns j in {2y, 2y+1} --------------------------
  __syncthreads();
  for (int idx = tid; idx < 2 * BATCH; idx += THREADS) {
    const int jj = idx >> 8, m = idx & 255;
    sall[idx] = x[m * INDIM + 2 * y + jj];
  }
  __syncthreads();
  for (int jj = 0; jj < 2; ++jj) {
    const float* cur = W2 + (size_t)n * (INDIM * INDIM) + (2 * y + jj) * INDIM;
    do_block(cur, jj == 0 ? cur + INDIM : nullptr, xb, sall + jj * BATCH,
             lane, acc);
  }

  // ---- degree 1: slice 0 only ---------------------------------------------
  if (y == 0) {
    __syncthreads();
    for (int m = tid; m < BATCH; m += THREADS) sall[m] = 1.0f;
    __syncthreads();
    do_block(W1 + (size_t)n * INDIM, nullptr, xb, sall, lane, acc);
  }

  // K slices combine via f32 atomics onto bias-initialized output.
  const int h = lane >> 4;
#pragma unroll
  for (int bt = 0; bt < 16; ++bt) {
#pragma unroll
    for (int v = 0; v < 8; ++v) {
      const int m = bt * 16 + v + 8 * h;      // D layout: M = v + 8h
      atomicAdd(&out[(size_t)m * OUTDIM + n], acc[bt][v]);
    }
  }
}

__global__ void pinet_init(const float* __restrict__ bias,
                           float* __restrict__ out) {
  const int idx = blockIdx.x * blockDim.x + threadIdx.x;
  if (idx < BATCH * OUTDIM) out[idx] = bias[idx & (OUTDIM - 1)];
}

extern "C" void kernel_launch(void* const* d_in, const int* in_sizes, int n_in,
                              void* d_out, int out_size, void* d_ws,
                              size_t ws_size, hipStream_t stream) {
  (void)in_sizes; (void)n_in; (void)d_ws; (void)ws_size; (void)out_size;
  const float* x  = (const float*)d_in[0];
  const float* W1 = (const float*)d_in[1];
  const float* W2 = (const float*)d_in[2];
  const float* W3 = (const float*)d_in[3];
  const float* b  = (const float*)d_in[4];
  float* out = (float*)d_out;

  pinet_init<<<dim3((BATCH * OUTDIM + 255) / 256), dim3(256), 0, stream>>>(b, out);
  pinet_main<<<dim3(OUTDIM / 64, KSLICES), dim3(THREADS), 0, stream>>>(
      x, W1, W2, W3, out);
}
